// Transformer_12704513261710
// MI455X (gfx1250) — matrix operand order repaired and checked by reading, not compiled, not run on hardware
//
#include <hip/hip_runtime.h>
#include <hip/hip_bf16.h>
#include <math.h>

// ---------------------------------------------------------------------------
// Types / helpers
// ---------------------------------------------------------------------------
typedef __attribute__((ext_vector_type(16))) __bf16 bf16x16;
typedef __attribute__((ext_vector_type(8)))  float  floatx8;
typedef __attribute__((ext_vector_type(4)))  unsigned int v4u;
typedef __attribute__((ext_vector_type(8)))  int          v8i;
typedef __attribute__((ext_vector_type(4)))  int          v4i;

union FragAB { uint4 q[2]; bf16x16 v; };

__device__ __forceinline__ unsigned short f2bf(float f) {
  unsigned u = __float_as_uint(f);
  u += 0x7FFFu + ((u >> 16) & 1u);           // round-to-nearest-even
  return (unsigned short)(u >> 16);
}

#define TBM 128
#define TBN 64
#define TBK 32
#define LDSS 40   // BK + 8 ushorts padding -> 80B row stride (16B aligned, bank-spread)

// ---------------------------------------------------------------------------
// TDM: async-DMA one bf16 tile [rows x TBK] (row stride in elements) from
// global into LDS at lds_byte_off, reproducing the padded LDS layout:
// 64B per row then 16B pad  (pad_interval=3 -> 16 DW, pad_amount=3 -> 4 DW).
// Tile must be fully in-bounds (tensor dims == tile dims). Issue from ONE wave.
// ---------------------------------------------------------------------------
__device__ __forceinline__ void tdm_load_tile_bf16(const unsigned short* gsrc,
                                                   unsigned lds_byte_off,
                                                   unsigned rows,
                                                   long long row_stride_elems) {
  unsigned long long ga = (unsigned long long)(uintptr_t)gsrc;
  v4u g0;
  g0[0] = 1u;                                            // count=1, user mode
  g0[1] = lds_byte_off;                                  // D#.lds_addr
  g0[2] = (unsigned)ga;                                  // global_addr[31:0]
  g0[3] = (unsigned)((ga >> 32) & 0x01FFFFFFull)         // global_addr[56:32]
        | (2u << 30);                                    // type = 2 ("image")
  const unsigned td0 = (unsigned)TBK;                    // row length (elems)
  const unsigned td1 = rows;
  const unsigned long long s0 = (unsigned long long)row_stride_elems;
  v8i g1;
  g1[0] = (int)((1u << 16)                               // data_size = 2 bytes
              | (1u << 20)                               // pad_enable
              | (3u << 22)                               // pad_interval: 16 DW
              | (3u << 25));                             // pad_amount:   4 DW
  g1[1] = (int)((td0 & 0xFFFFu) << 16);                  // tensor_dim0[15:0]
  g1[2] = (int)(((td0 >> 16) & 0xFFFFu) | ((td1 & 0xFFFFu) << 16));
  g1[3] = (int)(((td1 >> 16) & 0xFFFFu) | ((unsigned)TBK << 16)); // tile_dim0
  g1[4] = (int)(td1 & 0xFFFFu);                          // tile_dim1; tile_dim2=0
  g1[5] = (int)(unsigned)(s0 & 0xFFFFFFFFull);           // dim0_stride[31:0]
  g1[6] = (int)(unsigned)((s0 >> 32) & 0xFFFFull);       // dim0_stride[47:32]
  g1[7] = 0;                                             // dim1_stride unused (2D)
  v4i gz4 = {0, 0, 0, 0};
  v8i gz8 = {0, 0, 0, 0, 0, 0, 0, 0};
  __builtin_amdgcn_tensor_load_to_lds(g0, g1, gz4, gz4, gz8, 0);
}

// ---------------------------------------------------------------------------
// Generic batched GEMM:  C[bz] = act( alpha * A[bz] @ B[bz] + bias ) + residual
//   A: row-major (lda), fp32 or bf16 (bf16 A tiles DMA'd via TDM)
//   B: element (k,n) at b + k*ldb_k + n*ldb_n  (fp32 path assumes ldb_n==1;
//      bf16 transposed path (ldb_k==1) DMA'd via TDM)
// Tiling: 256 thr = 8 waves; BM=128, BN=64, BK=32; wave => 2x2 WMMA 16x16 tiles
// ---------------------------------------------------------------------------
struct GemmArgs {
  const void* a_ptr; const void* b_ptr; void* c_ptr;
  const float* bias; const float* residual;
  int M, N, K;
  int lda; long long ldb_k; long long ldb_n; int ldc;
  long long a_bs1, a_bs2, b_bs1, b_bs2, c_bs1, c_bs2;
  int heads;
  float alpha;
  int a_bf16, b_bf16, out_bf16, do_gelu;
};

__global__ __launch_bounds__(256) void gemm_wmma(GemmArgs g) {
  __shared__ alignas(16) unsigned short sA[TBM][LDSS];
  __shared__ alignas(16) unsigned short sB[TBN][LDSS];

  const int tid  = threadIdx.x;
  const int lane = tid & 31;
  const int wid  = tid >> 5;
  const int wm   = wid >> 1;     // 0..3 (M direction)
  const int wn   = wid & 1;      // 0..1 (N direction)
  const int lr   = lane & 15;
  const int hi   = lane >> 4;

  const int m0 = blockIdx.y * TBM;
  const int n0 = blockIdx.x * TBN;
  const int bz = blockIdx.z;
  int bb = 0, hh = 0;
  if (g.heads > 0) { bb = bz / g.heads; hh = bz % g.heads; }

  const long long aoff = (long long)bb * g.a_bs1 + (long long)hh * g.a_bs2;
  const long long boff = (long long)bb * g.b_bs1 + (long long)hh * g.b_bs2;
  const long long coff = (long long)bb * g.c_bs1 + (long long)hh * g.c_bs2;

  const unsigned sA_lds = (unsigned)(uintptr_t)(void*)&sA[0][0];
  const unsigned sB_lds = (unsigned)(uintptr_t)(void*)&sB[0][0];

  const bool b_tdm = (g.b_bf16 != 0) && (g.ldb_k == 1);

  floatx8 acc[2][2] = {};

  for (int k0 = 0; k0 < g.K; k0 += TBK) {
    // ---------------- load A tile ------------------------------------------
    if (!g.a_bf16) {                        // fp32 -> bf16 convert, cooperative
      const float* Ap = (const float*)g.a_ptr + aoff;
#pragma unroll
      for (int p = 0; p < 4; ++p) {
        int id  = tid + 256 * p;            // 1024 float4 groups
        int row = id >> 3;
        int kq  = (id & 7) << 2;
        float4 v = *(const float4*)(Ap + (long long)(m0 + row) * g.lda + (k0 + kq));
        unsigned short* d = &sA[row][kq];
        d[0] = f2bf(v.x); d[1] = f2bf(v.y); d[2] = f2bf(v.z); d[3] = f2bf(v.w);
      }
    } else if (wid == 0) {                  // bf16: single-wave TDM DMA
      const unsigned short* Ap = (const unsigned short*)g.a_ptr + aoff
                               + (long long)m0 * g.lda + k0;
      tdm_load_tile_bf16(Ap, sA_lds, TBM, g.lda);
    }
    // ---------------- load B tile transposed into sB[n][k] -----------------
    if (!g.b_bf16) {                        // fp32, row-major (ldb_n == 1)
      const float* Bp = (const float*)g.b_ptr + boff;
#pragma unroll
      for (int p = 0; p < 2; ++p) {
        int id = tid + 256 * p;             // 32 k-rows x 16 float4
        int kk = id >> 4;
        int nq = (id & 15) << 2;
        float4 v = make_float4(0.f, 0.f, 0.f, 0.f);
        if (n0 + nq < g.N) {
          const float* src = Bp + (long long)(k0 + kk) * g.ldb_k + (n0 + nq);
          v = *(const float4*)src;
          if (p == 0 && k0 + TBK < g.K)     // prefetch next K panel
            __builtin_prefetch(src + (long long)TBK * g.ldb_k, 0, 3);
        }
        sB[nq + 0][kk] = f2bf(v.x); sB[nq + 1][kk] = f2bf(v.y);
        sB[nq + 2][kk] = f2bf(v.z); sB[nq + 3][kk] = f2bf(v.w);
      }
    } else if (b_tdm) {                     // bf16 K^T tile: single-wave TDM DMA
      if (wid == 1) {
        const unsigned short* Bp = (const unsigned short*)g.b_ptr + boff
                                 + (long long)n0 * g.ldb_n + k0;
        tdm_load_tile_bf16(Bp, sB_lds, TBN, g.ldb_n);
      }
    } else {                                // bf16, row-major: transpose in LDS
      const unsigned short* Bp = (const unsigned short*)g.b_ptr + boff;
      int kk = tid >> 3;                    // 32 k x 8 groups of 8n
      int ng = (tid & 7) << 3;
      uint4 v = make_uint4(0, 0, 0, 0);
      if (n0 + ng < g.N)
        v = *(const uint4*)(Bp + (long long)(k0 + kk) * g.ldb_k + (n0 + ng));
      union { uint4 u; unsigned short s[8]; } t; t.u = v;
#pragma unroll
      for (int i = 0; i < 8; ++i) sB[ng + i][kk] = t.s[i];
    }
    // TDM issuers drain their TENSORcnt before the workgroup barrier
    if ((g.a_bf16 && wid == 0) || (b_tdm && wid == 1))
      __builtin_amdgcn_s_wait_tensorcnt(0);
    __syncthreads();

    // ---------------- fragments per ISA 7.12.2 layouts ---------------------
    FragAB afr[2], bfr[2];
#pragma unroll
    for (int tm = 0; tm < 2; ++tm) {        // A: M=lane&15; K = 8*hi+{0..7,16..23}
      int m = wm * 32 + tm * 16 + lr;
      afr[tm].q[0] = *(const uint4*)&sA[m][hi * 8];
      afr[tm].q[1] = *(const uint4*)&sA[m][hi * 8 + 16];
    }
#pragma unroll
    for (int tn = 0; tn < 2; ++tn) {        // B: N=lane&15; K = 8*hi+{0..7,16..23}
      int n = wn * 32 + tn * 16 + lr;
      bfr[tn].q[0] = *(const uint4*)&sB[n][hi * 8];
      bfr[tn].q[1] = *(const uint4*)&sB[n][hi * 8 + 16];
    }
#pragma unroll
    for (int tm = 0; tm < 2; ++tm)
#pragma unroll
      for (int tn = 0; tn < 2; ++tn)
        acc[tm][tn] = __builtin_amdgcn_wmma_f32_16x16x32_bf16(
            false, afr[tm].v, false, bfr[tn].v, (short)0, acc[tm][tn], false, false);
    __syncthreads();
  }

  // ---------------- epilogue: C layout M = r + 8*hi, N = lane&15 -----------
#pragma unroll
  for (int tm = 0; tm < 2; ++tm) {
#pragma unroll
    for (int tn = 0; tn < 2; ++tn) {
      int n = n0 + wn * 32 + tn * 16 + lr;
      if (n >= g.N) continue;
      float bv = g.bias ? g.bias[n] : 0.f;
#pragma unroll
      for (int r = 0; r < 8; ++r) {
        int m = m0 + wm * 32 + tm * 16 + hi * 8 + r;
        float v = acc[tm][tn][r] * g.alpha + bv;
        if (g.do_gelu) v = 0.5f * v * (1.f + erff(v * 0.70710678118654752f));
        if (g.residual) v += g.residual[(long long)m * g.ldc + n];
        long long ci = coff + (long long)m * g.ldc + n;
        if (g.out_bf16) ((unsigned short*)g.c_ptr)[ci] = f2bf(v);
        else            ((float*)g.c_ptr)[ci] = v;
      }
    }
  }
}

// ---------------------------------------------------------------------------
// LayerNorm (elementwise_affine=False) * (gamma + 1), D = 96, wave per row
// ---------------------------------------------------------------------------
__global__ __launch_bounds__(256) void ln_kernel(const float* __restrict__ x,
                                                 const float* __restrict__ gamma,
                                                 float* __restrict__ o, int rows) {
  int row  = (blockIdx.x * 256 + threadIdx.x) >> 5;
  int lane = threadIdx.x & 31;
  if (row >= rows) return;
  const float* r = x + (long long)row * 96;
  float v0 = r[lane], v1 = r[lane + 32], v2 = r[lane + 64];
  float s = v0 + v1 + v2;
#pragma unroll
  for (int off = 16; off; off >>= 1) s += __shfl_xor(s, off, 32);
  float mu = s * (1.f / 96.f);
  float d0 = v0 - mu, d1 = v1 - mu, d2 = v2 - mu;
  float q = d0 * d0 + d1 * d1 + d2 * d2;
#pragma unroll
  for (int off = 16; off; off >>= 1) q += __shfl_xor(q, off, 32);
  float inv = rsqrtf(q * (1.f / 96.f) + 1e-5f);
  float* w = o + (long long)row * 96;
  w[lane]      = d0 * inv * (gamma[lane]      + 1.f);
  w[lane + 32] = d1 * inv * (gamma[lane + 32] + 1.f);
  w[lane + 64] = d2 * inv * (gamma[lane + 64] + 1.f);
}

// ---------------------------------------------------------------------------
// Row softmax over 512 cols, fp32 in -> bf16 out, wave per row
// ---------------------------------------------------------------------------
__global__ __launch_bounds__(256) void softmax_kernel(const float* __restrict__ sim,
                                                      unsigned short* __restrict__ a,
                                                      int rows) {
  int row  = (blockIdx.x * 256 + threadIdx.x) >> 5;
  int lane = threadIdx.x & 31;
  if (row >= rows) return;
  const float* r = sim + (long long)row * 512;
  float v[16];
  float mx = -3.4e38f;
#pragma unroll
  for (int i = 0; i < 16; ++i) { v[i] = r[lane + 32 * i]; mx = fmaxf(mx, v[i]); }
#pragma unroll
  for (int off = 16; off; off >>= 1) mx = fmaxf(mx, __shfl_xor(mx, off, 32));
  float s = 0.f;
#pragma unroll
  for (int i = 0; i < 16; ++i) { v[i] = __expf(v[i] - mx); s += v[i]; }
#pragma unroll
  for (int off = 16; off; off >>= 1) s += __shfl_xor(s, off, 32);
  float inv = 1.f / s;
  unsigned short* w = a + (long long)row * 512;
#pragma unroll
  for (int i = 0; i < 16; ++i) w[lane + 32 * i] = f2bf(v[i] * inv);
}

// ---------------------------------------------------------------------------
// Host orchestration
// ---------------------------------------------------------------------------
extern "C" void kernel_launch(void* const* d_in, const int* in_sizes, int n_in,
                              void* d_out, int out_size, void* d_ws, size_t ws_size,
                              hipStream_t stream) {
  (void)in_sizes; (void)n_in; (void)out_size; (void)ws_size;
  const float* x_in    = (const float*)d_in[0];
  const float* gam_a   = (const float*)d_in[1];
  const float* Wq      = (const float*)d_in[2];
  const float* Wk      = (const float*)d_in[3];
  const float* Wv      = (const float*)d_in[4];
  const float* Wo      = (const float*)d_in[5];
  const float* gam_m   = (const float*)d_in[6];
  const float* W1      = (const float*)d_in[7];
  const float* b1      = (const float*)d_in[8];
  const float* W2      = (const float*)d_in[9];
  const float* b2      = (const float*)d_in[10];
  const float* gam_mid = (const float*)d_in[11];
  const float* gam_fin = (const float*)d_in[12];
  float* out = (float*)d_out;

  const int ROWS = 1024;                    // b*n = 2*512
  char* w = (char*)d_ws;
  auto alloc = [&](size_t bytes) {
    char* p = w; w += (bytes + 255) & ~(size_t)255; return p;
  };
  float*          h   = (float*)         alloc((size_t)ROWS * 96 * 4);
  unsigned short* qb  = (unsigned short*)alloc((size_t)ROWS * 6144 * 2);
  unsigned short* kb  = (unsigned short*)alloc((size_t)ROWS * 6144 * 2);
  unsigned short* vb  = (unsigned short*)alloc((size_t)ROWS * 6144 * 2);
  float*          sim = (float*)         alloc((size_t)16 * 512 * 512 * 4);
  unsigned short* ab  = (unsigned short*)alloc((size_t)16 * 512 * 512 * 2);
  unsigned short* ob  = (unsigned short*)alloc((size_t)ROWS * 6144 * 2);
  unsigned short* h1  = (unsigned short*)alloc((size_t)ROWS * 384 * 2);
  float*          xa  = (float*)         alloc((size_t)ROWS * 96 * 4);
  float*          xb  = (float*)         alloc((size_t)ROWS * 96 * 4);

  (void)hipMemcpyAsync(xa, x_in, (size_t)ROWS * 96 * 4,
                       hipMemcpyDeviceToDevice, stream);

  float* cur = xa;
  float* nxt = xb;

  auto do_gemm = [&](const GemmArgs& g, int nb) {
    dim3 grid((g.N + TBN - 1) / TBN, (g.M + TBM - 1) / TBM, nb);
    gemm_wmma<<<grid, 256, 0, stream>>>(g);
  };
  auto do_ln = [&](const float* src, const float* gamma, float* dst) {
    ln_kernel<<<ROWS / 8, 256, 0, stream>>>(src, gamma, dst, ROWS);
  };

  auto attn = [&](int li) {                 // li = d*3 + s
    const float* wq = Wq + (long long)li * 96 * 6144;
    const float* wk = Wk + (long long)li * 96 * 6144;
    const float* wv = Wv + (long long)li * 96 * 6144;
    const float* wo = Wo + (long long)li * 6144 * 96;
    do_ln(cur, gam_a + li * 96, h);

    GemmArgs g{};                           // h @ W{q,k,v}  -> bf16
    g.a_ptr = h; g.M = ROWS; g.N = 6144; g.K = 96;
    g.lda = 96; g.ldb_k = 6144; g.ldb_n = 1; g.ldc = 6144;
    g.heads = 1; g.alpha = 1.f; g.out_bf16 = 1;
    g.b_ptr = wq; g.c_ptr = qb; do_gemm(g, 1);
    g.b_ptr = wk; g.c_ptr = kb; do_gemm(g, 1);
    g.b_ptr = wv; g.c_ptr = vb; do_gemm(g, 1);

    GemmArgs s{};                           // sim = (q*scale) @ k^T  per (b,head)
    s.a_ptr = qb; s.b_ptr = kb; s.c_ptr = sim;
    s.M = 512; s.N = 512; s.K = 768;
    s.lda = 6144; s.ldb_k = 1; s.ldb_n = 6144; s.ldc = 512;
    s.a_bs1 = 512LL * 6144; s.a_bs2 = 768;
    s.b_bs1 = 512LL * 6144; s.b_bs2 = 768;
    s.c_bs1 = 8LL * 512 * 512; s.c_bs2 = 512LL * 512;
    s.heads = 8; s.alpha = 0.125f;          // 64^-0.5
    s.a_bf16 = 1; s.b_bf16 = 1;
    do_gemm(s, 16);

    softmax_kernel<<<16 * 512 / 8, 256, 0, stream>>>(sim, ab, 16 * 512);

    GemmArgs av{};                          // o = A @ V  per (b,head) -> bf16
    av.a_ptr = ab; av.b_ptr = vb; av.c_ptr = ob;
    av.M = 512; av.N = 768; av.K = 512;
    av.lda = 512; av.ldb_k = 6144; av.ldb_n = 1; av.ldc = 6144;
    av.a_bs1 = 8LL * 512 * 512; av.a_bs2 = 512LL * 512;
    av.b_bs1 = 512LL * 6144;    av.b_bs2 = 768;
    av.c_bs1 = 512LL * 6144;    av.c_bs2 = 768;
    av.heads = 8; av.alpha = 1.f; av.a_bf16 = 1; av.b_bf16 = 1; av.out_bf16 = 1;
    do_gemm(av, 16);

    GemmArgs o{};                           // x' = o @ Wo + x
    o.a_ptr = ob; o.b_ptr = wo; o.c_ptr = nxt;
    o.M = ROWS; o.N = 96; o.K = 6144;
    o.lda = 6144; o.ldb_k = 96; o.ldb_n = 1; o.ldc = 96;
    o.heads = 1; o.alpha = 1.f; o.a_bf16 = 1; o.residual = cur;
    do_gemm(o, 1);

    float* t = cur; cur = nxt; nxt = t;
  };

  auto mlp = [&](int li) {                  // li = d*2 + s
    do_ln(cur, gam_m + li * 96, h);
    GemmArgs g1{};                          // h1 = gelu(h @ W1 + b1) -> bf16
    g1.a_ptr = h; g1.b_ptr = W1 + (long long)li * 96 * 384; g1.c_ptr = h1;
    g1.bias = b1 + li * 384;
    g1.M = ROWS; g1.N = 384; g1.K = 96;
    g1.lda = 96; g1.ldb_k = 384; g1.ldb_n = 1; g1.ldc = 384;
    g1.heads = 1; g1.alpha = 1.f; g1.out_bf16 = 1; g1.do_gelu = 1;
    do_gemm(g1, 1);
    GemmArgs g2{};                          // x' = h1 @ W2 + b2 + x
    g2.a_ptr = h1; g2.b_ptr = W2 + (long long)li * 384 * 96; g2.c_ptr = nxt;
    g2.bias = b2 + li * 96; g2.residual = cur;
    g2.M = ROWS; g2.N = 96; g2.K = 384;
    g2.lda = 384; g2.ldb_k = 96; g2.ldb_n = 1; g2.ldc = 96;
    g2.heads = 1; g2.alpha = 1.f; g2.a_bf16 = 1;
    do_gemm(g2, 1);
    float* t = cur; cur = nxt; nxt = t;
  };

  for (int d = 0; d < 5; ++d) {
    attn(d * 3 + 0);
    mlp(d * 2 + 0);
    attn(d * 3 + 1);
    attn(d * 3 + 2);
    do_ln(cur, gam_mid + d * 96, nxt);      // bare LN, no residual
    { float* t = cur; cur = nxt; nxt = t; }
    mlp(d * 2 + 1);
  }
  ln_kernel<<<ROWS / 8, 256, 0, stream>>>(cur, gam_fin, out, ROWS);
}
